// DownUnit_33294586479032
// MI455X (gfx1250) — matrix operand order repaired and checked by reading, not compiled, not run on hardware
//
#include <hip/hip_runtime.h>
#include <hip/hip_bf16.h>

// ---------------- constants (match reference) ----------------
#define H_     1024
#define HEADS_ 16
#define DH_    64
#define W_     256
#define R_     128
#define I_     4096
#define S_     4096
#define NWIN_  31
#define B_     2
#define BN_    62
#define CHUNK_ 4          // windows per chunk (bounds workspace)
#define KSTEP_ 32

typedef __attribute__((ext_vector_type(16))) _Float16 v16h;
typedef __attribute__((ext_vector_type(8)))  _Float16 v8h;
typedef __attribute__((ext_vector_type(8)))  float    v8f;

__device__ __forceinline__ v8f v8f_zero() {
  v8f z = {0.f, 0.f, 0.f, 0.f, 0.f, 0.f, 0.f, 0.f};
  return z;
}

__device__ __forceinline__ v8f wmma32(v16h a, v16h b, v8f c) {
  return __builtin_amdgcn_wmma_f32_16x16x32_f16(
      /*neg_a=*/false, a, /*neg_b=*/false, b,
      /*c_mod=*/(short)0, c, /*reuse_a=*/false, /*reuse_b=*/false);
}

// ---- CDNA5 async global->LDS copy (ASYNCcnt-tracked) -----------------------
// LDS byte address = low 32 bits of generic pointer (ISA: aperture truncation).
__device__ __forceinline__ void async_copy_b128(void* lds_dst, const void* gsrc) {
#if defined(__gfx1250__)
  unsigned l = (unsigned)(unsigned long long)lds_dst;
  asm volatile("global_load_async_to_lds_b128 %0, %1, off"
               :: "v"(l), "v"(gsrc) : "memory");
#else
  *(v8h*)lds_dst = *(const v8h*)gsrc;
#endif
}

__device__ __forceinline__ void wait_async0() {
#if defined(__gfx1250__)
  asm volatile("s_wait_asynccnt 0x0" ::: "memory");
#endif
}

// A fragment: 16x32 (MxK) f16, row-major source with leading dim lda.
// ISA layout: lane m=L%16, hi=L/16; halves j: k = k0 + hi*8 + j (+16 for j>=8).
__device__ __forceinline__ v16h load_frag_a(const _Float16* __restrict__ base,
                                            int lda, int m0, int k0, int lane) {
  int m  = m0 + (lane & 15);
  int hi = lane >> 4;
  const _Float16* p = base + (size_t)m * lda + k0 + hi * 8;
  v8h lo = *(const v8h*)(p);
  v8h hh = *(const v8h*)(p + 16);
  v16h r;
#pragma unroll
  for (int i = 0; i < 8; ++i) { r[i] = lo[i]; r[i + 8] = hh[i]; }
  return r;
}

// B fragment: 32x16 (KxN) f16 where source is stored [N][K] row-major (ldb=K).
// ISA layout: lane n=L%16, hi=L/16; halves j: k = k0 + hi*8 + j (+16 for j>=8).
__device__ __forceinline__ v16h load_frag_b_nk(const _Float16* __restrict__ Wt,
                                               int ldb, int n0, int k0, int lane) {
  int n  = n0 + (lane & 15);
  int hi = lane >> 4;
  const _Float16* p = Wt + (size_t)n * ldb + k0 + hi * 8;
  v8h lo = *(const v8h*)(p);
  v8h hh = *(const v8h*)(p + 16);
  v16h r;
#pragma unroll
  for (int i = 0; i < 8; ++i) { r[i] = lo[i]; r[i + 8] = hh[i]; }
  return r;
}

// Map a chunk-local row m (= wloc*W + r) to the element offset of its source
// row in x[B,S,H]:  ((b*S + n*R + r) * H)
__device__ __forceinline__ size_t gather_row_off(int m, int c0) {
  int wloc = m >> 8;
  int r    = m & (W_ - 1);
  int wi   = c0 + wloc;
  int b    = wi / NWIN_;
  int n    = wi - b * NWIN_;
  return ((size_t)(b * S_ + n * R_ + r)) * H_;
}

// ---------------- fp32 -> fp16 weight conversion ----------------
__global__ void f32_to_f16_kernel(const float* __restrict__ s,
                                  _Float16* __restrict__ d, int n) {
  int i = blockIdx.x * 256 + threadIdx.x;
  if (i < n) d[i] = (_Float16)s[i];
}

// ---------------- RMSNorm (optionally gathering windows from x) ----------------
__global__ __launch_bounds__(256) void rmsnorm_kernel(
    const float* __restrict__ src, const float* __restrict__ w,
    _Float16* __restrict__ dst, int gather, int c0) {
  __shared__ float red[256];
  int m = blockIdx.x;
  int t = threadIdx.x;
  const float* row = gather ? (src + gather_row_off(m, c0))
                            : (src + (size_t)m * H_);
  float vals[4];
  float ss = 0.f;
#pragma unroll
  for (int i = 0; i < 4; ++i) {
    float v = row[t + i * 256];
    vals[i] = v;
    ss += v * v;
  }
  red[t] = ss;
  __syncthreads();
  for (int o = 128; o; o >>= 1) {
    if (t < o) red[t] += red[t + o];
    __syncthreads();
  }
  float rs = rsqrtf(red[0] * (1.f / (float)H_) + 1e-5f);
#pragma unroll
  for (int i = 0; i < 4; ++i) {
    int col = t + i * 256;
    dst[(size_t)m * H_ + col] = (_Float16)(vals[i] * rs * w[col]);
  }
}

// ---------------- LDS-staged, async double-buffered WMMA GEMM ----------------
// C[m,n] = sum_k A[m,k] * Bw[n,k]   (Bw stored [N][K] like the reference W)
// 256 threads = 8 waves in a 4(M) x 2(N) grid, 32x32 per wave -> 128x64 per block.
// Per K-step: async-stage A panel 128x32 + B panel 64x32 into LDS (double
// buffered), then ds_load fragments + 4 WMMAs per wave.
__global__ __launch_bounds__(256) void wmma_gemm_kernel(
    const _Float16* __restrict__ A, const _Float16* __restrict__ Bw,
    int M, int N, int K,
    float* __restrict__ outF32, _Float16* __restrict__ outF16,
    const float* __restrict__ resid, int resid_mode, int c0) {
  __shared__ __align__(16) _Float16 a_sh[2][128 * KSTEP_];  // 2 x 8 KB
  __shared__ __align__(16) _Float16 b_sh[2][64 * KSTEP_];   // 2 x 4 KB

  int tid = threadIdx.x;
  int lane = tid & 31;
  int wave = tid >> 5;
  int wm = wave & 3, wn = wave >> 2;
  int bm0 = blockIdx.x * 128;
  int bn0 = blockIdx.y * 64;

  int seg = tid & 3;        // 16B segment within a 32-half row
  int rowc = tid >> 2;      // 0..63

  v8f acc00 = v8f_zero(), acc01 = v8f_zero();
  v8f acc10 = v8f_zero(), acc11 = v8f_zero();

  const int steps = K / KSTEP_;

  // stage K-panel 0 into buffer 0
  {
    const int k0 = 0;
    async_copy_b128(&a_sh[0][rowc * KSTEP_ + seg * 8],
                    A + (size_t)(bm0 + rowc) * K + k0 + seg * 8);
    async_copy_b128(&a_sh[0][(rowc + 64) * KSTEP_ + seg * 8],
                    A + (size_t)(bm0 + rowc + 64) * K + k0 + seg * 8);
    async_copy_b128(&b_sh[0][rowc * KSTEP_ + seg * 8],
                    Bw + (size_t)(bn0 + rowc) * K + k0 + seg * 8);
  }

  for (int kt = 0; kt < steps; ++kt) {
    wait_async0();        // own copies for buffer kt&1 have landed (in-order)
    __syncthreads();      // everyone's copies landed; prior reads finished

    if (kt + 1 < steps) { // overlap next panel's fill with this panel's WMMAs
      int nb = (kt + 1) & 1;
      int k0 = (kt + 1) * KSTEP_;
      async_copy_b128(&a_sh[nb][rowc * KSTEP_ + seg * 8],
                      A + (size_t)(bm0 + rowc) * K + k0 + seg * 8);
      async_copy_b128(&a_sh[nb][(rowc + 64) * KSTEP_ + seg * 8],
                      A + (size_t)(bm0 + rowc + 64) * K + k0 + seg * 8);
      async_copy_b128(&b_sh[nb][rowc * KSTEP_ + seg * 8],
                      Bw + (size_t)(bn0 + rowc) * K + k0 + seg * 8);
    }

    const _Float16* ap = &a_sh[kt & 1][0];
    const _Float16* bp = &b_sh[kt & 1][0];
    v16h a0 = load_frag_a(ap, KSTEP_, wm * 32, 0, lane);
    v16h a1 = load_frag_a(ap, KSTEP_, wm * 32 + 16, 0, lane);
    v16h b0 = load_frag_b_nk(bp, KSTEP_, wn * 32, 0, lane);
    v16h b1 = load_frag_b_nk(bp, KSTEP_, wn * 32 + 16, 0, lane);
    acc00 = wmma32(a0, b0, acc00);
    acc01 = wmma32(a0, b1, acc01);
    acc10 = wmma32(a1, b0, acc10);
    acc11 = wmma32(a1, b1, acc11);
  }

  int hi = lane >> 4, nn = lane & 15;
  int m0 = bm0 + wm * 32;
  int n0 = bn0 + wn * 32;
  v8f accs[2][2] = {{acc00, acc01}, {acc10, acc11}};
#pragma unroll
  for (int i = 0; i < 2; ++i) {
#pragma unroll
    for (int j = 0; j < 2; ++j) {
#pragma unroll
      for (int r = 0; r < 8; ++r) {
        int row = m0 + i * 16 + r + hi * 8;
        int col = n0 + j * 16 + nn;
        float v = accs[i][j][r];
        if (resid_mode == 1) {
          v += resid[(size_t)row * N + col];
        } else if (resid_mode == 2) {
          v += resid[gather_row_off(row, c0) + col];
        }
        if (outF16) outF16[(size_t)row * N + col] = (_Float16)v;
        else        outF32[(size_t)row * N + col] = v;
      }
    }
  }
}

// ---------------- RoPE + head-major repack (fp16) ----------------
__global__ void rope_kernel(const _Float16* __restrict__ qkv,
                            const int* __restrict__ pos_ids,
                            _Float16* __restrict__ qo, _Float16* __restrict__ ko,
                            _Float16* __restrict__ vo, int cM, int c0) {
  int idx = blockIdx.x * 256 + threadIdx.x;
  if (idx >= cM * HEADS_ * 32) return;
  int i = idx & 31;
  int h = (idx >> 5) & (HEADS_ - 1);
  int m = idx >> 9;
  int wloc = m >> 8;
  int r = m & (W_ - 1);
  int wi = c0 + wloc;
  int b = wi / NWIN_;
  int n = wi - b * NWIN_;
  int pos = pos_ids[b * S_ + n * R_ + r];

  float invf = __expf(-((float)i / 32.f) * 9.210340372f);  // 10000^(-2i/64)
  float fr = (float)pos * invf;
  float cs = __cosf(fr), sn = __sinf(fr);

  size_t base = (size_t)m * (3 * H_) + h * DH_;
  size_t ob = ((size_t)((wloc * HEADS_ + h) * W_ + r)) * DH_;

  float q0 = (float)qkv[base + i],        q1 = (float)qkv[base + i + 32];
  float k0 = (float)qkv[base + H_ + i],   k1 = (float)qkv[base + H_ + i + 32];
  qo[ob + i]      = (_Float16)(q0 * cs - q1 * sn);
  qo[ob + i + 32] = (_Float16)(q1 * cs + q0 * sn);
  ko[ob + i]      = (_Float16)(k0 * cs - k1 * sn);
  ko[ob + i + 32] = (_Float16)(k1 * cs + k0 * sn);
  vo[ob + i]      = qkv[base + 2 * H_ + i];
  vo[ob + i + 32] = qkv[base + 2 * H_ + i + 32];
}

// ---------------- WMMA windowed causal attention ----------------
// One block = 2 waves; each wave owns one 16-row query tile of one (window, head).
#define AT_WAVES 2
__global__ __launch_bounds__(64) void attn_kernel(
    const _Float16* __restrict__ qv, const _Float16* __restrict__ kv,
    const _Float16* __restrict__ vv, _Float16* __restrict__ aout) {
  __shared__ __align__(16) _Float16 vt[DH_ * W_];              // V^T : 32 KB
  __shared__ __align__(16) _Float16 pstrip[AT_WAVES][16 * W_]; // P rows: 16 KB
  int tid = threadIdx.x;
  int lane = tid & 31;
  int wave = tid >> 5;
  int qgrp = blockIdx.x & 7;
  int wh = blockIdx.x >> 3;
  int h = wh & (HEADS_ - 1);
  int wloc = wh >> 4;

  size_t hb = (size_t)(wloc * HEADS_ + h) * W_ * DH_;
  const _Float16* qb = qv + hb;
  const _Float16* kb = kv + hb;
  const _Float16* vb = vv + hb;

  // cooperative transpose of V into LDS: vt[d][key]
  for (int e = tid; e < W_ * DH_; e += 64) {
    int key = e >> 6;
    int d = e & 63;
    vt[d * W_ + key] = vb[e];
  }
  __syncthreads();

  int qt = qgrp * AT_WAVES + wave;
  v16h aq0 = load_frag_a(qb, DH_, qt * 16, 0, lane);
  v16h aq1 = load_frag_a(qb, DH_, qt * 16, 32, lane);

  int hi = lane >> 4, nn = lane & 15;
  v8f sacc[16];
#pragma unroll
  for (int t = 0; t < 16; ++t) {
    v8f c = v8f_zero();
    v16h b0 = load_frag_b_nk(kb, DH_, t * 16, 0, lane);
    v16h b1 = load_frag_b_nk(kb, DH_, t * 16, 32, lane);
    c = wmma32(aq0, b0, c);
    c = wmma32(aq1, b1, c);
#pragma unroll
    for (int r = 0; r < 8; ++r) {
      int row = qt * 16 + r + hi * 8;
      int col = t * 16 + nn;
      float s = c[r] * 0.125f;  // 1/sqrt(64)
      c[r] = (col > row) ? -1.0e30f : s;
    }
    sacc[t] = c;
  }

  // row softmax: reduce over 16 tiles (register) x 16 lanes (half-wave shuffles)
  float inv_[8];
#pragma unroll
  for (int r = 0; r < 8; ++r) {
    float mx = -3.0e38f;
#pragma unroll
    for (int t = 0; t < 16; ++t) mx = fmaxf(mx, sacc[t][r]);
    mx = fmaxf(mx, __shfl_xor(mx, 1, 32));
    mx = fmaxf(mx, __shfl_xor(mx, 2, 32));
    mx = fmaxf(mx, __shfl_xor(mx, 4, 32));
    mx = fmaxf(mx, __shfl_xor(mx, 8, 32));
    float sm = 0.f;
#pragma unroll
    for (int t = 0; t < 16; ++t) {
      float p = __expf(sacc[t][r] - mx);
      sacc[t][r] = p;
      sm += p;
    }
    sm += __shfl_xor(sm, 1, 32);
    sm += __shfl_xor(sm, 2, 32);
    sm += __shfl_xor(sm, 4, 32);
    sm += __shfl_xor(sm, 8, 32);
    inv_[r] = 1.f / sm;
  }

  // stage normalized P (f16) in LDS, row-major 16 x 256 per wave
  _Float16* ps = &pstrip[wave][0];
#pragma unroll
  for (int t = 0; t < 16; ++t) {
#pragma unroll
    for (int r = 0; r < 8; ++r) {
      ps[(r + hi * 8) * W_ + t * 16 + nn] = (_Float16)(sacc[t][r] * inv_[r]);
    }
  }

  // O = P @ V via WMMA (A from LDS P-strip, B from LDS V^T)
#pragma unroll
  for (int n0 = 0; n0 < DH_; n0 += 16) {
    v8f oc = v8f_zero();
#pragma unroll
    for (int kt = 0; kt < 8; ++kt) {
      v16h ap = load_frag_a(ps, W_, 0, kt * 32, lane);
      v16h bv = load_frag_b_nk(vt, W_, n0, kt * 32, lane);
      oc = wmma32(ap, bv, oc);
    }
#pragma unroll
    for (int r = 0; r < 8; ++r) {
      int mrow = wloc * W_ + qt * 16 + r + hi * 8;
      int col = h * DH_ + n0 + nn;
      aout[(size_t)mrow * H_ + col] = (_Float16)oc[r];
    }
  }
}

// ---------------- SwiGLU ----------------
__global__ void swiglu_kernel(const _Float16* __restrict__ gu,
                              _Float16* __restrict__ act, int cM) {
  int idx = blockIdx.x * 256 + threadIdx.x;
  if (idx >= cM * I_) return;
  int m = idx >> 12;
  int i = idx & (I_ - 1);
  float g = (float)gu[(size_t)m * (2 * I_) + i];
  float u = (float)gu[(size_t)m * (2 * I_) + I_ + i];
  float sg = g / (1.f + __expf(-g));
  act[idx] = (_Float16)(u * sg);
}

// ---------------- attention-pool per window ----------------
__global__ __launch_bounds__(256) void pool_kernel(
    const float* __restrict__ yfull, const float* __restrict__ scw,
    float* __restrict__ z) {
  __shared__ float sc[256];
  __shared__ float tmp[256];
  int wi = blockIdx.x;  // 0..BN-1 (b-major, matches reference reshape)
  int r = threadIdx.x;
  const float* yrow = yfull + ((size_t)(wi * W_ + r)) * H_;
  float s = 0.f;
  for (int j = 0; j < H_; ++j) s += yrow[j] * scw[j];
  tmp[r] = s;
  __syncthreads();
  for (int o = 128; o; o >>= 1) {
    if (r < o) tmp[r] = fmaxf(tmp[r], tmp[r + o]);
    __syncthreads();
  }
  float mx = tmp[0];
  __syncthreads();
  float e = __expf(s - mx);
  sc[r] = e;
  tmp[r] = e;
  __syncthreads();
  for (int o = 128; o; o >>= 1) {
    if (r < o) tmp[r] += tmp[r + o];
    __syncthreads();
  }
  float inv = 1.f / tmp[0];
  __syncthreads();
#pragma unroll
  for (int cb = 0; cb < 4; ++cb) {
    int col = r * 4 + cb;
    float acc = 0.f;
    for (int rr = 0; rr < W_; ++rr)
      acc += sc[rr] * yfull[((size_t)(wi * W_ + rr)) * H_ + col];
    z[(size_t)wi * H_ + col] = acc * inv;
  }
}

// ---------------- deterministic triangular overlap-add stitch ----------------
__global__ void stitch_kernel(const float* __restrict__ yfull,
                              float* __restrict__ yl) {
  size_t idx = (size_t)blockIdx.x * 256 + threadIdx.x;  // < B*S*H
  int col = (int)(idx & (H_ - 1));
  int p = (int)((idx >> 10) & (S_ - 1));
  int b = (int)(idx >> 22);
  int nlo = (p <= W_ - 1) ? 0 : ((p - (W_ - 1) + (R_ - 1)) >> 7);
  int nhi = p >> 7;
  if (nhi > NWIN_ - 1) nhi = NWIN_ - 1;
  float acc = 0.f, wsum = 0.f;
  for (int n = nlo; n <= nhi; ++n) {
    int r = p - n * R_;
    int mn = (r < W_ - r) ? r : (W_ - r);
    float blend = (float)mn * (1.f / 128.f);
    acc += yfull[((size_t)((b * NWIN_ + n) * W_ + r)) * H_ + col] * blend;
    wsum += blend;
  }
  yl[idx] = acc / (wsum + 1e-8f);
}

__global__ void blockpos_kernel(int* __restrict__ bp) {
  int i = threadIdx.x;
  if (i < B_ * NWIN_) bp[i] = (i % NWIN_) * R_ + (W_ - 1);
}

// ---------------- host orchestration ----------------
extern "C" void kernel_launch(void* const* d_in, const int* in_sizes, int n_in,
                              void* d_out, int out_size, void* d_ws, size_t ws_size,
                              hipStream_t stream) {
  (void)in_sizes; (void)n_in; (void)out_size; (void)ws_size;
  const float* x    = (const float*)d_in[0];
  const int*   pos  = (const int*)d_in[1];
  const float* innw = (const float*)d_in[2];
  const float* mlpw = (const float*)d_in[3];
  const float* qkvw = (const float*)d_in[4];
  const float* ow   = (const float*)d_in[5];
  const float* guw  = (const float*)d_in[6];
  const float* dww  = (const float*)d_in[7];
  const float* scw  = (const float*)d_in[8];

  char* ws = (char*)d_ws;
  size_t off = 0;
  auto alloc = [&](size_t bytes) -> char* {
    char* p = ws + off;
    off += (bytes + 255) & ~(size_t)255;
    return p;
  };

  // persistent
  float*    yfull   = (float*)alloc((size_t)BN_ * W_ * H_ * 4);       // 65 MB
  _Float16* qkvw_h  = (_Float16*)alloc((size_t)3 * H_ * H_ * 2);
  _Float16* ow_h    = (_Float16*)alloc((size_t)H_ * H_ * 2);
  _Float16* guw_h   = (_Float16*)alloc((size_t)2 * I_ * H_ * 2);
  _Float16* dww_h   = (_Float16*)alloc((size_t)H_ * I_ * 2);
  // per-chunk scratch (sized for CHUNK_ windows)
  const int MC = CHUNK_ * W_;  // 1024 rows
  _Float16* ynorm = (_Float16*)alloc((size_t)MC * H_ * 2);
  _Float16* qkvf  = (_Float16*)alloc((size_t)MC * 3 * H_ * 2);
  _Float16* qf    = (_Float16*)alloc((size_t)MC * H_ * 2);
  _Float16* kf    = (_Float16*)alloc((size_t)MC * H_ * 2);
  _Float16* vf    = (_Float16*)alloc((size_t)MC * H_ * 2);
  _Float16* attnf = (_Float16*)alloc((size_t)MC * H_ * 2);
  float*    ychk  = (float*)alloc((size_t)MC * H_ * 4);
  _Float16* y2f   = (_Float16*)alloc((size_t)MC * H_ * 2);
  _Float16* guf   = (_Float16*)alloc((size_t)MC * 2 * I_ * 2);
  _Float16* actf  = (_Float16*)alloc((size_t)MC * I_ * 2);

  // outputs
  float* ylocal = (float*)d_out;
  float* zout   = ylocal + (size_t)B_ * S_ * H_;
  int*   bpout  = (int*)(zout + (size_t)B_ * NWIN_ * H_);

  // ---- convert weights to fp16 (L2-resident for all GEMMs) ----
  {
    int n;
    n = 3 * H_ * H_;  f32_to_f16_kernel<<<(n + 255) / 256, 256, 0, stream>>>(qkvw, qkvw_h, n);
    n = H_ * H_;      f32_to_f16_kernel<<<(n + 255) / 256, 256, 0, stream>>>(ow, ow_h, n);
    n = 2 * I_ * H_;  f32_to_f16_kernel<<<(n + 255) / 256, 256, 0, stream>>>(guw, guw_h, n);
    n = H_ * I_;      f32_to_f16_kernel<<<(n + 255) / 256, 256, 0, stream>>>(dww, dww_h, n);
  }

  // ---- chunked window pipeline ----
  for (int c0 = 0; c0 < BN_; c0 += CHUNK_) {
    int cw = (BN_ - c0 < CHUNK_) ? (BN_ - c0) : CHUNK_;
    int cM = cw * W_;

    rmsnorm_kernel<<<cM, 256, 0, stream>>>(x, innw, ynorm, /*gather=*/1, c0);

    wmma_gemm_kernel<<<dim3(cM / 128, (3 * H_) / 64), 256, 0, stream>>>(
        ynorm, qkvw_h, cM, 3 * H_, H_, nullptr, qkvf, nullptr, 0, c0);

    rope_kernel<<<(cM * HEADS_ * 32 + 255) / 256, 256, 0, stream>>>(
        qkvf, pos, qf, kf, vf, cM, c0);

    attn_kernel<<<cw * HEADS_ * (16 / AT_WAVES), 64, 0, stream>>>(qf, kf, vf, attnf);

    wmma_gemm_kernel<<<dim3(cM / 128, H_ / 64), 256, 0, stream>>>(
        attnf, ow_h, cM, H_, H_, ychk, nullptr, x, /*resid gather x*/ 2, c0);

    rmsnorm_kernel<<<cM, 256, 0, stream>>>(ychk, mlpw, y2f, /*gather=*/0, c0);

    wmma_gemm_kernel<<<dim3(cM / 128, (2 * I_) / 64), 256, 0, stream>>>(
        y2f, guw_h, cM, 2 * I_, H_, nullptr, guf, nullptr, 0, c0);

    swiglu_kernel<<<(cM * I_ + 255) / 256, 256, 0, stream>>>(guf, actf, cM);

    wmma_gemm_kernel<<<dim3(cM / 128, H_ / 64), 256, 0, stream>>>(
        actf, dww_h, cM, H_, I_, yfull + (size_t)c0 * W_ * H_, nullptr,
        ychk, /*resid plain*/ 1, c0);
  }

  // ---- pooled summaries, stitch, positions ----
  pool_kernel<<<BN_, 256, 0, stream>>>(yfull, scw, zout);
  stitch_kernel<<<(B_ * S_ * H_) / 256, 256, 0, stream>>>(yfull, ylocal);
  blockpos_kernel<<<1, 64, 0, stream>>>(bpout);
}
